// GAT_75496935129296
// MI455X (gfx1250) — matrix operand order repaired and checked by reading, not compiled, not run on hardware
//
#include <hip/hip_runtime.h>

#define N_NODES 4096
#define IN_FEAT 512
#define OUT_FEAT 64
#define HEADS 4
#define HDIM (OUT_FEAT * HEADS) /* 256 */
#define ALPHA 0.2f
#define NEG_INF_V (-9.0e15f)

typedef __attribute__((ext_vector_type(16))) __bf16        v16bf;
typedef __attribute__((ext_vector_type(8)))  float         v8f;
typedef __attribute__((ext_vector_type(8)))  unsigned int  v8u;

union BF16Tile { v16bf bf; v8u u; };

__device__ __forceinline__ unsigned int f2bf(float f) {
  unsigned int u = __float_as_uint(f);
  return (u + 0x7FFFu + ((u >> 16) & 1u)) >> 16; // RNE
}
__device__ __forceinline__ float bf2f(unsigned short b) {
  return __uint_as_float(((unsigned int)b) << 16);
}
// A-operand (16x32 bf16) K position for VGPR v, lane-half h (ISA 7.12.2 layout)
__device__ __forceinline__ int kposA(int v, int half) {
  return ((v >> 2) << 4) + half * 8 + ((v & 3) << 1);
}

// ---------------- Kernel 0a: x (f32) -> x_bf16 row-major, packed pairs -------
__global__ __launch_bounds__(256) void cvt_x_kernel(const float* __restrict__ x,
                                                    unsigned int* __restrict__ xbf_u32) {
  int gid = blockIdx.x * 256 + threadIdx.x;          // 1,048,576 pairs
  float2 v = ((const float2*)x)[gid];
  xbf_u32[gid] = f2bf(v.x) | (f2bf(v.y) << 16);
}

// ---------------- Kernel 0b: W (512x256 f32) -> W^T bf16 (256x512) ----------
__global__ __launch_bounds__(256) void cvt_wt_kernel(const float* __restrict__ W,
                                                     unsigned short* __restrict__ wt) {
  int gid = blockIdx.x * 256 + threadIdx.x;          // 131,072
  int n = gid >> 9;                                   // 0..255
  int k = gid & 511;                                  // 0..511
  wt[gid] = (unsigned short)f2bf(W[k * HDIM + n]);
}

// ---------------- Kernel 1: H = x@W via WMMA; store H^T bf16 (256 x 4096) ---
__global__ __launch_bounds__(128) void gemm_h_kernel(const unsigned short* __restrict__ xbf,
                                                     const unsigned short* __restrict__ wtbf,
                                                     unsigned short* __restrict__ hT) {
  const int lane = threadIdx.x & 31;
  const int wave = threadIdx.x >> 5;
  const int i0 = blockIdx.x * 16;                     // row tile
  const int c0 = (blockIdx.y * 4 + wave) * 16;        // col tile (0..255)
  const int M = lane & 15;
  const int half = lane >> 4;

  v8f acc = {};
  const unsigned short* arow = xbf + (size_t)(i0 + M) * IN_FEAT;
  const unsigned short* brow = wtbf + (size_t)(c0 + M) * IN_FEAT; // N == lane&15

  for (int k0 = 0; k0 < IN_FEAT; k0 += 32) {
    BF16Tile a, b;
#pragma unroll
    for (int v = 0; v < 8; ++v) {
      a.u[v] = *(const unsigned int*)(arow + k0 + kposA(v, half));
      b.u[v] = *(const unsigned int*)(brow + k0 + kposA(v, half));
    }
    acc = __builtin_amdgcn_wmma_f32_16x16x32_bf16(false, a.bf, false, b.bf,
                                                  (short)0, acc, false, false);
  }
  // C layout: VGPR r holds row (r + 8*half), col = lane&15.
  // Lane's 8 values are 8 consecutive H-rows at one column -> one b128 store in H^T.
  unsigned int pk[4];
#pragma unroll
  for (int j = 0; j < 4; ++j)
    pk[j] = f2bf(acc[2 * j]) | (f2bf(acc[2 * j + 1]) << 16);
  size_t off = (size_t)(c0 + M) * N_NODES + i0 + 8 * half;
  *(uint4*)(hT + off) = make_uint4(pk[0], pk[1], pk[2], pk[3]);
}

// ---------------- Kernel 2: s,t per (node, head), SoA [head][n] -------------
__global__ __launch_bounds__(256) void st_kernel(const unsigned short* __restrict__ hT,
                                                 const float* __restrict__ a,
                                                 float* __restrict__ s_arr,
                                                 float* __restrict__ t_arr) {
  int gid = blockIdx.x * 256 + threadIdx.x;           // 16384
  int head = gid >> 12;
  int n = gid & (N_NODES - 1);
  float s = 0.f, t = 0.f;
#pragma unroll 8
  for (int f = 0; f < OUT_FEAT; ++f) {
    float hv = bf2f(hT[(size_t)(head * OUT_FEAT + f) * N_NODES + n]);
    s += hv * a[f];
    t += hv * a[OUT_FEAT + f];
  }
  s_arr[head * N_NODES + n] = s;
  t_arr[head * N_NODES + n] = t;
}

// ---------------- Kernel 3: fused masked softmax + aggregation (flash) ------
// One wave per head; 16 rows per block; j processed 64 at a time (2 A tiles
// per online-softmax step to amortize max/rescale/bpermute overhead).
__global__ __launch_bounds__(128) void gat_attn_kernel(const int* __restrict__ adj,
                                                       const unsigned short* __restrict__ hT,
                                                       const float* __restrict__ s_arr,
                                                       const float* __restrict__ t_arr,
                                                       float* __restrict__ out) {
  const int lane = threadIdx.x & 31;
  const int head = threadIdx.x >> 5;                  // one wave per head
  const int i0 = blockIdx.x * 16;
  const int M = lane & 15;                            // row within tile
  const int half = lane >> 4;

  const float s_row = s_arr[head * N_NODES + i0 + M];
  const int* adj_row = adj + (size_t)(i0 + M) * N_NODES;
  const float* t_head = t_arr + head * N_NODES;

  float m = -__builtin_inff();
  float l = 0.f;
  v8f acc[4];
#pragma unroll
  for (int nt = 0; nt < 4; ++nt) acc[nt] = {};

  for (int j0 = 0; j0 < N_NODES; j0 += 64) {
    if (j0 + 64 < N_NODES)
      __builtin_prefetch(adj_row + j0 + 64, 0, 1);    // global_prefetch_b8

    // e values at exactly the A-operand K positions of this lane (2 tiles)
    float e[32];
#pragma unroll
    for (int t2 = 0; t2 < 2; ++t2) {
#pragma unroll
      for (int v = 0; v < 8; ++v) {
        int k = t2 * 32 + kposA(v, half);
        int2 ad = *(const int2*)(adj_row + j0 + k);
        float2 tt = *(const float2*)(t_head + j0 + k);
        float r0 = s_row + tt.x; r0 = fmaxf(r0, ALPHA * r0);
        float r1 = s_row + tt.y; r1 = fmaxf(r1, ALPHA * r1);
        e[t2 * 16 + 2 * v]     = (ad.x > 0) ? r0 : NEG_INF_V;
        e[t2 * 16 + 2 * v + 1] = (ad.y > 0) ? r1 : NEG_INF_V;
      }
    }
    // online softmax: row i lives in lanes M and M+16
    float vmax = e[0];
#pragma unroll
    for (int q = 1; q < 32; ++q) vmax = fmaxf(vmax, e[q]);
    vmax = fmaxf(vmax, __shfl_xor(vmax, 16, 32));
    float mnew = fmaxf(m, vmax);
    float cfac = __expf(m - mnew);

    BF16Tile aop[2];
    float psum = 0.f;
#pragma unroll
    for (int t2 = 0; t2 < 2; ++t2) {
#pragma unroll
      for (int v = 0; v < 8; ++v) {
        float p0 = __expf(e[t2 * 16 + 2 * v] - mnew);
        float p1 = __expf(e[t2 * 16 + 2 * v + 1] - mnew);
        psum += p0 + p1;
        aop[t2].u[v] = f2bf(p0) | (f2bf(p1) << 16);
      }
    }
    psum += __shfl_xor(psum, 16, 32);
    l = l * cfac + psum;
    m = mnew;

    // rescale accumulators only when some row's max advanced (wave-uniform
    // branch => EXEC stays all-ones inside: safe for bpermute and WMMA)
    if (__any(cfac < 1.0f)) {
#pragma unroll
      for (int r = 0; r < 8; ++r) {
        float cr = __shfl(cfac, r + 8 * half, 32);
#pragma unroll
        for (int nt = 0; nt < 4; ++nt) acc[nt][r] *= cr;
      }
    }

    // acc += P(16x64) @ H(64x64) for this head, as 2 K-tiles x 4 N-tiles
#pragma unroll
    for (int t2 = 0; t2 < 2; ++t2) {
#pragma unroll
      for (int nt = 0; nt < 4; ++nt) {
        BF16Tile bop;
        const unsigned short* bcol =
            hT + (size_t)(head * OUT_FEAT + nt * 16 + M) * N_NODES; // N == lane&15
#pragma unroll
        for (int v = 0; v < 8; ++v)
          bop.u[v] = *(const unsigned int*)(bcol + j0 + t2 * 32 + kposA(v, half));
        acc[nt] = __builtin_amdgcn_wmma_f32_16x16x32_bf16(false, aop[t2].bf, false, bop.bf,
                                                          (short)0, acc[nt], false, false);
      }
    }
  }

  // normalize by row sum and write fp32 output (N x 256, head-major cols)
#pragma unroll
  for (int r = 0; r < 8; ++r) {
    float lr = __shfl(l, r + 8 * half, 32);
    float inv = 1.0f / lr;
    int row = i0 + r + 8 * half;
#pragma unroll
    for (int nt = 0; nt < 4; ++nt)
      out[(size_t)row * HDIM + head * OUT_FEAT + nt * 16 + M] = acc[nt][r] * inv;
  }
}

// ---------------------------------------------------------------------------
extern "C" void kernel_launch(void* const* d_in, const int* in_sizes, int n_in,
                              void* d_out, int out_size, void* d_ws, size_t ws_size,
                              hipStream_t stream) {
  (void)in_sizes; (void)n_in; (void)out_size; (void)ws_size;
  const float* x   = (const float*)d_in[0];
  const int*   adj = (const int*)d_in[1];
  const float* W   = (const float*)d_in[2];
  const float* a   = (const float*)d_in[3];
  float* out = (float*)d_out;

  char* ws = (char*)d_ws;
  unsigned short* xbf  = (unsigned short*)(ws);                       // 4 MB
  unsigned short* wtbf = (unsigned short*)(ws + (4u << 20));          // 256 KB
  unsigned short* hT   = (unsigned short*)(ws + (4u << 20) + (256u << 10)); // 2 MB
  float* s_arr = (float*)(ws + (6u << 20) + (256u << 10));            // 64 KB
  float* t_arr = (float*)(ws + (6u << 20) + (320u << 10));            // 64 KB

  cvt_x_kernel<<<dim3((N_NODES * IN_FEAT / 2) / 256), dim3(256), 0, stream>>>(
      x, (unsigned int*)xbf);
  cvt_wt_kernel<<<dim3((HDIM * IN_FEAT) / 256), dim3(256), 0, stream>>>(W, wtbf);
  gemm_h_kernel<<<dim3(N_NODES / 16, 4), dim3(128), 0, stream>>>(xbf, wtbf, hT);
  st_kernel<<<dim3((N_NODES * HEADS) / 256), dim3(256), 0, stream>>>(hT, a, s_arr, t_arr);
  gat_attn_kernel<<<dim3(N_NODES / 16), dim3(128), 0, stream>>>(adj, hT, s_arr, t_arr, out);
}
